// GNN_24756191494623
// MI455X (gfx1250) — matrix operand order repaired and checked by reading, not compiled, not run on hardware
//
#include <hip/hip_runtime.h>
#include <hip/hip_bf16.h>

typedef __attribute__((ext_vector_type(16))) __bf16 v16bf;
typedef __attribute__((ext_vector_type(8)))  __bf16 v8bf;
typedef __attribute__((ext_vector_type(8)))  float  v8f;

#define NS_ 16384
#define NT_ 16384
#define NE_ 327680
#define NB_ 64

__device__ __forceinline__ float nan0(float v) { return isfinite(v) ? v : 0.0f; }

// Low 32 bits of a generic LDS pointer == wave-relative LDS byte address
// (CDNA5 ISA 10.2).
__device__ __forceinline__ uint32_t lds_lo(const void* p) {
  return (uint32_t)(uintptr_t)p;
}

// CDNA5 async global->LDS copy, 16B per lane, tracked by ASYNCcnt.
__device__ __forceinline__ void async_copy16(uint32_t lds_addr, const void* gp) {
  asm volatile("global_load_async_to_lds_b128 %0, %1, off"
               :: "v"(lds_addr), "v"((unsigned long long)(uintptr_t)gp)
               : "memory");
}
__device__ __forceinline__ void wait_async0() {
  asm volatile("s_wait_asynccnt 0" ::: "memory");
}

// =====================================================================
// WMMA bf16 GEMM:  C[M x Nout] = act(A[M x K] @ Wt^T + bias)
// Wt is PRE-TRANSPOSED: [BN rows][K cols], row-major, K multiple of 32.
// Block: 256 threads = 8 waves, tile 128 x BN. One wave -> 16 rows x BN.
// Double-buffered LDS staging via GLOBAL_LOAD_ASYNC_TO_LDS_B128.
// =====================================================================
template<int BN, int ACT>
__global__ __launch_bounds__(256)
void k_gemm(const __bf16* __restrict__ A, int lda,
            const __bf16* __restrict__ Wt,
            const float*  __restrict__ bias,
            float*  __restrict__ Cf, int ldcf,
            __bf16* __restrict__ Cb, int ldcb,
            int M, int K, int Nout)
{
  constexpr int NTL = BN / 16;
  __shared__ alignas(16) __bf16 ldsA[2][128 * 40];
  __shared__ alignas(16) __bf16 ldsW[2][BN * 40];
  const int tid  = threadIdx.x;
  const int wave = tid >> 5;
  const int lane = tid & 31;
  const int hi   = lane >> 4;     // 0: lanes 0-15, 1: lanes 16-31
  const int ln   = lane & 15;
  const long rowBase = (long)blockIdx.x * 128;

  v8f acc[NTL];
  #pragma unroll
  for (int t = 0; t < NTL; ++t)
    #pragma unroll
    for (int i = 0; i < 8; ++i) acc[t][i] = 0.0f;

  // Branchless async staging of one K-slab (A: 128x32, W: BNx32) into buf.
  auto stage = [&](int k0, int buf) {
    for (int c = tid; c < 512; c += 256) {
      int r  = c >> 2;
      int kk = (c & 3) * 8;
      long gr  = rowBase + r;
      long grc = (gr < (long)M) ? gr : 0;  // clamp; rows >= M are store-masked
      async_copy16(lds_lo(&ldsA[buf][r * 40 + kk]),
                   A + grc * (long)lda + k0 + kk);
    }
    for (int c = tid; c < BN * 4; c += 256) {
      int n  = c >> 2;
      int kk = (c & 3) * 8;
      async_copy16(lds_lo(&ldsW[buf][n * 40 + kk]),
                   Wt + (long)n * K + k0 + kk);
    }
  };

  stage(0, 0);
  int buf = 0;

  for (int k0 = 0; k0 < K; k0 += 32) {
    wait_async0();       // my async writes for this slab are in LDS
    __syncthreads();     // everyone's are
    if (k0 + 32 < K) stage(k0 + 32, buf ^ 1);  // prefetch next slab

    // A fragment per ISA 16-bit 16x32 layout:
    //  half j (0..7)  -> K = j   + hi*8 ; half j (8..15) -> K = j+8 + hi*8
    const int ar = wave * 16 + ln;
    v8bf alo = *(const v8bf*)(&ldsA[buf][ar * 40 + hi * 8]);
    v8bf ahi = *(const v8bf*)(&ldsA[buf][ar * 40 + 16 + hi * 8]);
    v16bf af;
    #pragma unroll
    for (int i = 0; i < 8; ++i) { af[i] = alo[i]; af[i + 8] = ahi[i]; }

    #pragma unroll
    for (int t = 0; t < NTL; ++t) {
      // B fragment: half j (0..7) -> K = j + hi*8; half j (8..15) -> K = j+8 + hi*8; N = t*16 + ln
      const int bn = t * 16 + ln;
      v8bf blo = *(const v8bf*)(&ldsW[buf][bn * 40 + hi * 8]);
      v8bf bhi = *(const v8bf*)(&ldsW[buf][bn * 40 + 16 + hi * 8]);
      v16bf bfr;
      #pragma unroll
      for (int i = 0; i < 8; ++i) { bfr[i] = blo[i]; bfr[i + 8] = bhi[i]; }
      acc[t] = __builtin_amdgcn_wmma_f32_16x16x32_bf16(
          false, af, false, bfr, (short)0, acc[t], false, false);
    }
    buf ^= 1;
  }

  // Epilogue: C layout: VGPR r, lanes 0-15 -> M=r, lanes 16-31 -> M=r+8
  #pragma unroll
  for (int t = 0; t < NTL; ++t) {
    const int col = t * 16 + ln;
    const float bv = bias[col];
    #pragma unroll
    for (int r = 0; r < 8; ++r) {
      long row = rowBase + wave * 16 + r + hi * 8;
      if (row < (long)M && col < Nout) {
        float v = acc[t][r] + bv;
        if (ACT) v = v > 0.0f ? v : 0.1f * v;
        if (Cf) Cf[row * (long)ldcf + col] = v;
        if (Cb) Cb[row * (long)ldcb + col] = (__bf16)v;
      }
    }
  }
}

// ============ weight conversion: W[K][N] f32 -> Wt[Np][Kp] bf16 (padded) ====
__global__ void k_wt(const float* __restrict__ W, const float* __restrict__ b,
                     __bf16* __restrict__ Wt, float* __restrict__ bp,
                     int K, int N, int Kp, int Np)
{
  long total = (long)Np * Kp;
  for (long i = (long)blockIdx.x * blockDim.x + threadIdx.x; i < total;
       i += (long)gridDim.x * blockDim.x) {
    int n = (int)(i / Kp);
    int k = (int)(i - (long)n * Kp);
    float v = (k < K && n < N) ? W[(long)k * N + n] : 0.0f;
    Wt[i] = (__bf16)v;
    if (k == 0) bp[n] = (n < N) ? b[n] : 0.0f;
  }
}

// ==================== concat / gather kernels (f32 -> bf16) =================
__global__ void k_cat_edge(const float* __restrict__ xs, const float* __restrict__ xt,
                           const float* __restrict__ ea, const float* __restrict__ u,
                           const int* __restrict__ src, const int* __restrict__ tgt,
                           const int* __restrict__ be, __bf16* __restrict__ out)
{
  const long total = (long)NE_ * 224;
  for (long i = (long)blockIdx.x * blockDim.x + threadIdx.x; i < total;
       i += (long)gridDim.x * blockDim.x) {
    long e = i / 224; int j = (int)(i - e * 224);
    float v;
    if      (j < 64)  v = xs[(long)src[e] * 64 + j];
    else if (j < 128) v = xt[(long)tgt[e] * 64 + (j - 64)];
    else if (j < 192) v = ea[e * 64 + (j - 128)];
    else              v = u[(long)be[e] * 32 + (j - 192)];
    out[i] = (__bf16)v;
  }
}

__global__ void k_cat_ne(const float* __restrict__ node, const int* __restrict__ idx,
                         const float* __restrict__ ef, __bf16* __restrict__ out)
{
  const long total = (long)NE_ * 128;
  for (long i = (long)blockIdx.x * blockDim.x + threadIdx.x; i < total;
       i += (long)gridDim.x * blockDim.x) {
    long e = i >> 7; int j = (int)(i & 127);
    float v = (j < 64) ? node[(long)idx[e] * 64 + j] : ef[e * 64 + (j - 64)];
    out[i] = (__bf16)v;
  }
}

__global__ void k_cat_s2(const float* __restrict__ xs, const float* __restrict__ cnt,
                         const float* __restrict__ meanb, const float* __restrict__ stdb,
                         const float* __restrict__ skewb, const float* __restrict__ kurtb,
                         const float* __restrict__ u, const int* __restrict__ bs,
                         __bf16* __restrict__ out)
{
  const long total = (long)NS_ * 640;
  for (long i = (long)blockIdx.x * blockDim.x + threadIdx.x; i < total;
       i += (long)gridDim.x * blockDim.x) {
    long n = i / 640; int j = (int)(i - n * 640);
    float v;
    if      (j < 64)  v = xs[n * 64 + j];
    else if (j == 64) v = cnt[n];
    else if (j < 193) v = meanb[n * 128 + (j - 65)];
    else if (j < 321) v = stdb [n * 128 + (j - 193)];
    else if (j < 449) v = skewb[n * 128 + (j - 321)];
    else if (j < 577) v = kurtb[n * 128 + (j - 449)];
    else if (j < 609) v = u[(long)bs[n] * 32 + (j - 577)];
    else              v = 0.0f;
    out[i] = (__bf16)v;
  }
}

__global__ void k_cat_t2(const float* __restrict__ xt, const float* __restrict__ agg,
                         const float* __restrict__ u, const int* __restrict__ bt,
                         __bf16* __restrict__ out)
{
  const long total = (long)NT_ * 224;
  for (long i = (long)blockIdx.x * blockDim.x + threadIdx.x; i < total;
       i += (long)gridDim.x * blockDim.x) {
    long n = i / 224; int j = (int)(i - n * 224);
    float v;
    if      (j < 64)  v = xt[n * 64 + j];
    else if (j < 192) v = agg[n * 128 + (j - 64)];
    else              v = u[(long)bt[n] * 32 + (j - 192)];
    out[i] = (__bf16)v;
  }
}

__global__ void k_cat_g(const float* __restrict__ u, const float* __restrict__ sm,
                        const float* __restrict__ tm, __bf16* __restrict__ out)
{
  const long total = (long)NB_ * 160;
  for (long i = (long)blockIdx.x * blockDim.x + threadIdx.x; i < total;
       i += (long)gridDim.x * blockDim.x) {
    long b = i / 160; int j = (int)(i - b * 160);
    float v;
    if      (j < 32) v = u[b * 32 + j];
    else if (j < 96) v = sm[b * 64 + (j - 32)];
    else             v = tm[b * 64 + (j - 96)];
    out[i] = (__bf16)v;
  }
}

// ==================== segment moment statistics =============================
__global__ void k_stats1(const float* __restrict__ msg, const int* __restrict__ src,
                         float* __restrict__ cnt, float* __restrict__ sum,
                         float* __restrict__ sq)
{
  const long total = (long)NE_ * 128;
  for (long i = (long)blockIdx.x * blockDim.x + threadIdx.x; i < total;
       i += (long)gridDim.x * blockDim.x) {
    long e = i >> 7; int c = (int)(i & 127);
    float v = msg[i];
    long o = (long)src[e] * 128 + c;
    atomicAdd(&sum[o], v);
    atomicAdd(&sq[o], v * v);
    if (c == 0) atomicAdd(&cnt[src[e]], 1.0f);
  }
}

__global__ void k_fin1(const float* __restrict__ cnt, float* __restrict__ sum,
                       const float* __restrict__ sq, float* __restrict__ stdb)
{
  const long total = (long)NS_ * 128;
  for (long i = (long)blockIdx.x * blockDim.x + threadIdx.x; i < total;
       i += (long)gridDim.x * blockDim.x) {
    long n = i >> 7;
    float cc  = fmaxf(cnt[n], 1.0f);
    float m   = nan0(sum[i] / cc);
    float var = nan0(fmaxf(sq[i] / cc - m * m, 0.0f));
    sum[i]  = m;
    stdb[i] = sqrtf(var + 1e-6f);
  }
}

__global__ void k_stats2(const float* __restrict__ msg, const int* __restrict__ src,
                         const float* __restrict__ meanb,
                         float* __restrict__ s3, float* __restrict__ s4)
{
  const long total = (long)NE_ * 128;
  for (long i = (long)blockIdx.x * blockDim.x + threadIdx.x; i < total;
       i += (long)gridDim.x * blockDim.x) {
    long e = i >> 7; int c = (int)(i & 127);
    long o = (long)src[e] * 128 + c;
    float d = msg[i] - meanb[o];
    float d2 = d * d;
    atomicAdd(&s3[o], d2 * d);
    atomicAdd(&s4[o], d2 * d2);
  }
}

__global__ void k_fin2(const float* __restrict__ cnt, const float* __restrict__ stdb,
                       float* __restrict__ s3, float* __restrict__ s4)
{
  const long total = (long)NS_ * 128;
  for (long i = (long)blockIdx.x * blockDim.x + threadIdx.x; i < total;
       i += (long)gridDim.x * blockDim.x) {
    long n = i >> 7;
    float cc = fmaxf(cnt[n], 1.0f);
    float st = stdb[i];
    float st3 = st * st * st;
    s3[i] = nan0(s3[i] / (cc * st3));
    s4[i] = nan0(s4[i] / (cc * st3 * st));
  }
}

__global__ void k_scatter_add(const float* __restrict__ msg, const int* __restrict__ tgt,
                              float* __restrict__ agg)
{
  const long total = (long)NE_ * 128;
  for (long i = (long)blockIdx.x * blockDim.x + threadIdx.x; i < total;
       i += (long)gridDim.x * blockDim.x) {
    long e = i >> 7; int c = (int)(i & 127);
    atomicAdd(&agg[(long)tgt[e] * 128 + c], msg[i]);
  }
}

// ==================== graph-level means =====================================
__global__ void k_graph_sum(const float* __restrict__ x, const int* __restrict__ batch,
                            float* __restrict__ gsum, float* __restrict__ gcnt, long M)
{
  const long total = M * 64;
  for (long i = (long)blockIdx.x * blockDim.x + threadIdx.x; i < total;
       i += (long)gridDim.x * blockDim.x) {
    long n = i >> 6; int c = (int)(i & 63);
    atomicAdd(&gsum[(long)batch[n] * 64 + c], x[i]);
    if (c == 0) atomicAdd(&gcnt[batch[n]], 1.0f);
  }
}

__global__ void k_graph_mean(float* __restrict__ gsum, const float* __restrict__ gcnt)
{
  long i = (long)blockIdx.x * blockDim.x + threadIdx.x;
  if (i < (long)NB_ * 64) gsum[i] /= fmaxf(gcnt[i >> 6], 1.0f);
}

// ==================== batch-norm (per column over M rows) ===================
__global__ void k_colstats(const float* __restrict__ x, long M,
                           float* __restrict__ cs, float* __restrict__ cq)
{
  __shared__ float ssum[256], ssq[256];
  int t = threadIdx.x; int c = t & 63; int rr = t >> 6;
  float s = 0.0f, q = 0.0f;
  for (long r = (long)blockIdx.x * 4 + rr; r < M; r += (long)gridDim.x * 4) {
    float v = x[r * 64 + c];
    s += v; q += v * v;
  }
  ssum[t] = s; ssq[t] = q;
  __syncthreads();
  if (t < 64) {
    s = ssum[t] + ssum[t + 64] + ssum[t + 128] + ssum[t + 192];
    q = ssq[t]  + ssq[t + 64]  + ssq[t + 128]  + ssq[t + 192];
    atomicAdd(&cs[c], s);
    atomicAdd(&cq[c], q);
  }
}

__global__ void k_bn_apply(float* __restrict__ x, long M,
                           const float* __restrict__ cs, const float* __restrict__ cq)
{
  const float inv = 1.0f / (float)M;
  const long total = M * 64;
  for (long i = (long)blockIdx.x * blockDim.x + threadIdx.x; i < total;
       i += (long)gridDim.x * blockDim.x) {
    int c = (int)(i & 63);
    float m = cs[c] * inv;
    float var = cq[c] * inv - m * m;
    x[i] = (x[i] - m) * rsqrtf(var + 1e-5f);
  }
}

// ==================== softmax expectation over 50 classes ===================
__global__ void k_time(const float* __restrict__ logits, float* __restrict__ out)
{
  long e = (long)blockIdx.x * blockDim.x + threadIdx.x;
  if (e >= NE_) return;
  const float* l = logits + e * 64;
  float mx = l[0];
  for (int j = 1; j < 50; ++j) mx = fmaxf(mx, l[j]);
  float s = 0.0f, a = 0.0f;
  for (int j = 0; j < 50; ++j) {
    float p = expf(l[j] - mx);
    s += p; a += p * (float)j;
  }
  out[e] = a / s;
}

// ===========================================================================
// Host orchestration
// ===========================================================================
struct MW { __bf16* w1; float* b1; __bf16* w2; float* b2; };
struct MDim { int K1, N1, K1p, N1p, K2, N2, K2p, N2p; };

static const MDim MD[7] = {
  {224, 64, 224, 64,   64, 64, 64, 64},    // edge
  {128, 128, 128, 128, 128, 128, 128, 128},// s1
  {609, 64, 640, 64,   64, 64, 64, 64},    // s2
  {128, 128, 128, 128, 128, 128, 128, 128},// t1
  {224, 64, 224, 64,   64, 64, 64, 64},    // t2
  {160, 32, 160, 64,   32, 32, 32, 64},    // g
  {224, 50, 224, 64,   50, 50, 64, 64},    // last
};

static void gemm(hipStream_t s, const __bf16* A, int lda, const __bf16* Wt,
                 const float* bias, float* Cf, int ldcf, __bf16* Cb, int ldcb,
                 long M, int K, int Np, int Nout, int act)
{
  dim3 g((unsigned)((M + 127) / 128)), b(256);
  if (Np == 128) {
    if (act) k_gemm<128, 1><<<g, b, 0, s>>>(A, lda, Wt, bias, Cf, ldcf, Cb, ldcb, (int)M, K, Nout);
    else     k_gemm<128, 0><<<g, b, 0, s>>>(A, lda, Wt, bias, Cf, ldcf, Cb, ldcb, (int)M, K, Nout);
  } else {
    if (act) k_gemm<64, 1><<<g, b, 0, s>>>(A, lda, Wt, bias, Cf, ldcf, Cb, ldcb, (int)M, K, Nout);
    else     k_gemm<64, 0><<<g, b, 0, s>>>(A, lda, Wt, bias, Cf, ldcf, Cb, ldcb, (int)M, K, Nout);
  }
}

extern "C" void kernel_launch(void* const* d_in, const int* in_sizes, int n_in,
                              void* d_out, int out_size, void* d_ws, size_t ws_size,
                              hipStream_t stream)
{
  (void)in_sizes; (void)n_in; (void)out_size; (void)ws_size;
  const float* in_xs = (const float*)d_in[0];
  const float* in_xt = (const float*)d_in[1];
  const int*   src   = (const int*)d_in[2];
  const int*   tgt   = src + NE_;
  const float* in_ea = (const float*)d_in[3];
  const float* in_u  = (const float*)d_in[4];
  const int*   be    = (const int*)d_in[5];
  const int*   bs    = (const int*)d_in[6];
  const int*   bt    = (const int*)d_in[7];
  float* out = (float*)d_out;

  // ---- workspace allocator ----
  char* ws = (char*)d_ws;
  size_t cur = 0;
  auto alloc = [&](size_t bytes) -> void* {
    void* p = ws + cur;
    cur = (cur + bytes + 255) & ~(size_t)255;
    return p;
  };

  float*  xs_f  = (float*)alloc((size_t)NS_ * 64 * 4);
  float*  xt_f  = (float*)alloc((size_t)NT_ * 64 * 4);
  float*  u_f   = (float*)alloc((size_t)NB_ * 32 * 4);
  float*  Ef32  = (float*)alloc((size_t)NE_ * 64 * 4);
  __bf16* bufA  = (__bf16*)alloc((size_t)NE_ * 224 * 2);
  __bf16* Hf    = (__bf16*)alloc((size_t)NE_ * 128 * 2);
  float*  Msg   = (float*)alloc((size_t)NE_ * 128 * 4);
  float*  cnt   = (float*)alloc((size_t)NS_ * 4);
  float*  meanb = (float*)alloc((size_t)NS_ * 128 * 4);
  float*  sqb   = (float*)alloc((size_t)NS_ * 128 * 4);
  float*  stdb  = (float*)alloc((size_t)NS_ * 128 * 4);
  float*  skewb = (float*)alloc((size_t)NS_ * 128 * 4);
  float*  kurtb = (float*)alloc((size_t)NS_ * 128 * 4);
  float*  agg   = (float*)alloc((size_t)NT_ * 128 * 4);
  float*  gs_sum = (float*)alloc((size_t)NB_ * 64 * 4);
  float*  gt_sum = (float*)alloc((size_t)NB_ * 64 * 4);
  float*  gs_cnt = (float*)alloc((size_t)NB_ * 4);
  float*  gt_cnt = (float*)alloc((size_t)NB_ * 4);
  float*  colcs  = (float*)alloc(64 * 4);
  float*  colcq  = (float*)alloc(64 * 4);

  // ---- weights: convert f32 [K][N] -> bf16 transposed padded [Np][Kp] ----
  MW mw[25];
  for (int i = 0; i < 25; ++i) {
    int t = (i < 24) ? (i % 6) : 6;
    const MDim& d = MD[t];
    mw[i].w1 = (__bf16*)alloc((size_t)d.N1p * d.K1p * 2);
    mw[i].b1 = (float*) alloc((size_t)d.N1p * 4);
    mw[i].w2 = (__bf16*)alloc((size_t)d.N2p * d.K2p * 2);
    mw[i].b2 = (float*) alloc((size_t)d.N2p * 4);
  }
  for (int i = 0; i < 25; ++i) {
    int t  = (i < 24) ? (i % 6) : 6;
    int lb = 8 + i * 4;
    const MDim& d = MD[t];
    {
      long tot = (long)d.N1p * d.K1p;
      k_wt<<<dim3((unsigned)((tot + 255) / 256)), 256, 0, stream>>>(
          (const float*)d_in[lb + 0], (const float*)d_in[lb + 1],
          mw[i].w1, mw[i].b1, d.K1, d.N1, d.K1p, d.N1p);
    }
    {
      long tot = (long)d.N2p * d.K2p;
      k_wt<<<dim3((unsigned)((tot + 255) / 256)), 256, 0, stream>>>(
          (const float*)d_in[lb + 2], (const float*)d_in[lb + 3],
          mw[i].w2, mw[i].b2, d.K2, d.N2, d.K2p, d.N2p);
    }
  }

  // ---- initial state ----
  hipMemcpyAsync(xs_f, in_xs, (size_t)NS_ * 64 * 4, hipMemcpyDeviceToDevice, stream);
  hipMemcpyAsync(xt_f, in_xt, (size_t)NT_ * 64 * 4, hipMemcpyDeviceToDevice, stream);
  hipMemcpyAsync(u_f,  in_u,  (size_t)NB_ * 32 * 4, hipMemcpyDeviceToDevice, stream);
  hipMemcpyAsync(Ef32, in_ea, (size_t)NE_ * 64 * 4, hipMemcpyDeviceToDevice, stream);

  const dim3 GE(4096), GN(1024), B256(256);

  for (int b = 0; b < 4; ++b) {
    const MW* p = &mw[b * 6];
    const MDim& De = MD[0]; const MDim& Ds1 = MD[1]; const MDim& Ds2 = MD[2];
    const MDim& Dt1 = MD[3]; const MDim& Dt2 = MD[4]; const MDim& Dg = MD[5];

    // 1. edge MLP: cat(E x 224) -> h(E x 64) -> Ef32(E x 64)
    k_cat_edge<<<GE, B256, 0, stream>>>(xs_f, xt_f, Ef32, u_f, src, tgt, be, bufA);
    gemm(stream, bufA, 224, p[0].w1, p[0].b1, nullptr, 0, Hf, 64, NE_, De.K1p, De.N1p, De.N1p, 1);
    gemm(stream, Hf, 64, p[0].w2, p[0].b2, Ef32, 64, nullptr, 0, NE_, De.K2p, De.N2p, 64, 0);

    // 2. s1 MLP: cat(E x 128) -> h(E x 128) -> Msg(E x 128)
    k_cat_ne<<<GE, B256, 0, stream>>>(xt_f, tgt, Ef32, bufA);
    gemm(stream, bufA, 128, p[1].w1, p[1].b1, nullptr, 0, Hf, 128, NE_, Ds1.K1p, Ds1.N1p, Ds1.N1p, 1);
    gemm(stream, Hf, 128, p[1].w2, p[1].b2, Msg, 128, nullptr, 0, NE_, Ds1.K2p, Ds1.N2p, 128, 0);

    // 3. segment moments
    hipMemsetAsync(cnt,   0, (size_t)NS_ * 4,       stream);
    hipMemsetAsync(meanb, 0, (size_t)NS_ * 128 * 4, stream);
    hipMemsetAsync(sqb,   0, (size_t)NS_ * 128 * 4, stream);
    hipMemsetAsync(skewb, 0, (size_t)NS_ * 128 * 4, stream);
    hipMemsetAsync(kurtb, 0, (size_t)NS_ * 128 * 4, stream);
    k_stats1<<<GE, B256, 0, stream>>>(Msg, src, cnt, meanb, sqb);
    k_fin1<<<GN, B256, 0, stream>>>(cnt, meanb, sqb, stdb);
    k_stats2<<<GE, B256, 0, stream>>>(Msg, src, meanb, skewb, kurtb);
    k_fin2<<<GN, B256, 0, stream>>>(cnt, stdb, skewb, kurtb);

    // 4. s2 MLP: cat(NS x 640) -> h(NS x 64) -> xs_f(NS x 64)
    k_cat_s2<<<GN, B256, 0, stream>>>(xs_f, cnt, meanb, stdb, skewb, kurtb, u_f, bs, bufA);
    gemm(stream, bufA, 640, p[2].w1, p[2].b1, nullptr, 0, Hf, 64, NS_, Ds2.K1p, Ds2.N1p, Ds2.N1p, 1);
    gemm(stream, Hf, 64, p[2].w2, p[2].b2, xs_f, 64, nullptr, 0, NS_, Ds2.K2p, Ds2.N2p, 64, 0);

    // 5. t1 MLP: cat(E x 128) -> h -> Msg (msg_t)
    k_cat_ne<<<GE, B256, 0, stream>>>(xs_f, src, Ef32, bufA);
    gemm(stream, bufA, 128, p[3].w1, p[3].b1, nullptr, 0, Hf, 128, NE_, Dt1.K1p, Dt1.N1p, Dt1.N1p, 1);
    gemm(stream, Hf, 128, p[3].w2, p[3].b2, Msg, 128, nullptr, 0, NE_, Dt1.K2p, Dt1.N2p, 128, 0);

    // 6. aggregate to targets
    hipMemsetAsync(agg, 0, (size_t)NT_ * 128 * 4, stream);
    k_scatter_add<<<GE, B256, 0, stream>>>(Msg, tgt, agg);

    // 7. t2 MLP: cat(NT x 224) -> h -> xt_f
    k_cat_t2<<<GN, B256, 0, stream>>>(xt_f, agg, u_f, bt, bufA);
    gemm(stream, bufA, 224, p[4].w1, p[4].b1, nullptr, 0, Hf, 64, NT_, Dt2.K1p, Dt2.N1p, Dt2.N1p, 1);
    gemm(stream, Hf, 64, p[4].w2, p[4].b2, xt_f, 64, nullptr, 0, NT_, Dt2.K2p, Dt2.N2p, 64, 0);

    // 8. graph-level means
    hipMemsetAsync(gs_sum, 0, (size_t)NB_ * 64 * 4, stream);
    hipMemsetAsync(gt_sum, 0, (size_t)NB_ * 64 * 4, stream);
    hipMemsetAsync(gs_cnt, 0, (size_t)NB_ * 4, stream);
    hipMemsetAsync(gt_cnt, 0, (size_t)NB_ * 4, stream);
    k_graph_sum<<<GN, B256, 0, stream>>>(xs_f, bs, gs_sum, gs_cnt, NS_);
    k_graph_sum<<<GN, B256, 0, stream>>>(xt_f, bt, gt_sum, gt_cnt, NT_);
    k_graph_mean<<<dim3(16), B256, 0, stream>>>(gs_sum, gs_cnt);
    k_graph_mean<<<dim3(16), B256, 0, stream>>>(gt_sum, gt_cnt);

    // 9. global MLP: cat(B x 160) -> h(B x 32 padded 64) -> u_f(B x 32)
    k_cat_g<<<dim3(40), B256, 0, stream>>>(u_f, gs_sum, gt_sum, bufA);
    gemm(stream, bufA, 160, p[5].w1, p[5].b1, nullptr, 0, Hf, 64, NB_, Dg.K1p, Dg.N1p, Dg.N1p, 1);
    gemm(stream, Hf, 64, p[5].w2, p[5].b2, u_f, 32, nullptr, 0, NB_, Dg.K2p, Dg.N2p, 32, 0);

    // 10. batch norms (x_s, x_t, edge_attr)
    hipMemsetAsync(colcs, 0, 64 * 4, stream);
    hipMemsetAsync(colcq, 0, 64 * 4, stream);
    k_colstats<<<dim3(512), B256, 0, stream>>>(xs_f, NS_, colcs, colcq);
    k_bn_apply<<<GN, B256, 0, stream>>>(xs_f, NS_, colcs, colcq);

    hipMemsetAsync(colcs, 0, 64 * 4, stream);
    hipMemsetAsync(colcq, 0, 64 * 4, stream);
    k_colstats<<<dim3(512), B256, 0, stream>>>(xt_f, NT_, colcs, colcq);
    k_bn_apply<<<GN, B256, 0, stream>>>(xt_f, NT_, colcs, colcq);

    hipMemsetAsync(colcs, 0, 64 * 4, stream);
    hipMemsetAsync(colcq, 0, 64 * 4, stream);
    k_colstats<<<dim3(2048), B256, 0, stream>>>(Ef32, NE_, colcs, colcq);
    k_bn_apply<<<GE, B256, 0, stream>>>(Ef32, NE_, colcs, colcq);
  }

  // ---- final MLP + softmax expectation ----
  const MDim& Dl = MD[6];
  k_cat_edge<<<GE, B256, 0, stream>>>(xs_f, xt_f, Ef32, u_f, src, tgt, be, bufA);
  gemm(stream, bufA, 224, mw[24].w1, mw[24].b1, nullptr, 0, Hf, 64, NE_, Dl.K1p, Dl.N1p, Dl.N1p, 1);
  gemm(stream, Hf, 64, mw[24].w2, mw[24].b2, Msg, 64, nullptr, 0, NE_, Dl.K2p, Dl.N2p, 64, 0);
  k_time<<<dim3((NE_ + 255) / 256), B256, 0, stream>>>(Msg, out);
}
